// PointNet2MSGClassifier_53128745452259
// MI455X (gfx1250) — compile-verified, matrix operand order checked by reading, not run on hardware
//
#include <hip/hip_runtime.h>

typedef __attribute__((ext_vector_type(16))) _Float16 v16h;
typedef __attribute__((ext_vector_type(8)))  float    v8f;
typedef __attribute__((ext_vector_type(4)))  unsigned int v4u;
typedef __attribute__((ext_vector_type(8)))  int      v8i;
typedef __attribute__((ext_vector_type(4)))  int      v4i;

constexpr int   BSZ    = 8;
constexpr int   NPTS   = 4096;
constexpr float BN_EPS = 1e-5f;

struct LayerDesc { int wi, O, C; };   // wi = d_in index of W; b=wi+1, g=wi+2, be=wi+3

static __host__ __device__ inline int rnd32i(int c) { return (c + 31) & ~31; }

// ---------------------------------------------------------------------------
// TDM: issue a 2-D f16 tile load (32 cols x 64 rows) into LDS.
// LDS row pitch = 40 halfs (32 data + 8 pad) via pad_interval=3 (16 DWORDs),
// pad_amount=3 (4 DWORDs). Rows/cols beyond tensor_dim zero-fill / drop.
__device__ inline void tdm_load_tile_f16(unsigned lds_off, const _Float16* gptr,
                                         unsigned rows_rem, unsigned cols_rem,
                                         unsigned rowstride_elems) {
  unsigned long long ga = (unsigned long long)(uintptr_t)gptr;
  v4u g0;
  g0[0] = 1u;                                            // count=1, user mode
  g0[1] = lds_off;                                       // lds_addr (bytes)
  g0[2] = (unsigned)(ga & 0xFFFFFFFFull);                // global_addr[31:0]
  g0[3] = (unsigned)((ga >> 32) & 0x1FFFFFFull)          // global_addr[56:32]
        | (2u << 30);                                    // type = 2 (image)
  v8i g1;
  g1[0] = (int)((1u << 16)                               // data_size = 2B
              | (1u << 20)                               // pad_enable
              | (3u << 22)                               // pad_interval: 16 DW
              | (3u << 25));                             // pad_amount: 4 DW
  g1[1] = (int)((cols_rem & 0xFFFFu) << 16);             // tensor_dim0[15:0]
  g1[2] = (int)(((cols_rem >> 16) & 0xFFFFu)             // tensor_dim0[31:16]
              | ((rows_rem & 0xFFFFu) << 16));           // tensor_dim1[15:0]
  g1[3] = (int)(((rows_rem >> 16) & 0xFFFFu)             // tensor_dim1[31:16]
              | (32u << 16));                            // tile_dim0 = 32
  g1[4] = (int)64u;                                      // tile_dim1=64, tile_dim2=0
  g1[5] = (int)rowstride_elems;                          // tensor_dim0_stride[31:0]
  g1[6] = 0;                                             // stride hi / dim1_stride lo
  g1[7] = 0;
  v4i z4 = {0, 0, 0, 0};
#if defined(__clang_major__) && (__clang_major__ >= 23)
  v8i z8 = {0, 0, 0, 0, 0, 0, 0, 0};
  __builtin_amdgcn_tensor_load_to_lds(g0, g1, z4, z4, z8, 0);
#else
  __builtin_amdgcn_tensor_load_to_lds(g0, g1, z4, z4, 0);
#endif
}

// ---------------------------------------------------------------------------
// x [B,6,N] -> xyz [B,N,3], nrm [B,N,3]
__global__ void k_split_xyz(const float* __restrict__ x,
                            float* __restrict__ xyz, float* __restrict__ nrm) {
  int t = blockIdx.x * 256 + threadIdx.x;
  if (t >= BSZ * NPTS) return;
  int b = t / NPTS, n = t % NPTS;
  const float* xb = x + (size_t)b * 6 * NPTS;
#pragma unroll
  for (int c = 0; c < 3; ++c) {
    xyz[(size_t)t * 3 + c] = xb[(size_t)c * NPTS + n];
    nrm[(size_t)t * 3 + c] = xb[(size_t)(3 + c) * NPTS + n];
  }
}

// ---------------------------------------------------------------------------
// Farthest point sampling; one block (256 thr) per batch. N <= 4096.
__global__ void k_fps(const float* __restrict__ xyz, int N, int npoint,
                      int* __restrict__ fidx) {
  int b = blockIdx.x;
  const float* p = xyz + (size_t)b * N * 3;
  __shared__ float dist[4096];
  __shared__ float rv[256];
  __shared__ int   ri[256];
  __shared__ int   cur;
  int tid = threadIdx.x;
  for (int n = tid; n < N; n += 256) dist[n] = 1e10f;
  if (tid == 0) cur = 0;
  __syncthreads();
  for (int i = 0; i < npoint; ++i) {
    int c = cur;
    float cx = p[c * 3 + 0], cy = p[c * 3 + 1], cz = p[c * 3 + 2];
    if (tid == 0) fidx[b * npoint + i] = c;
    float best = -1.f; int bi = 0;
    for (int n = tid; n < N; n += 256) {
      float dx = p[n * 3 + 0] - cx;
      float dy = p[n * 3 + 1] - cy;
      float dz = p[n * 3 + 2] - cz;
      float d  = dx * dx + dy * dy + dz * dz;
      float dm = fminf(dist[n], d);
      dist[n] = dm;
      if (dm > best) { best = dm; bi = n; }
    }
    rv[tid] = best; ri[tid] = bi;
    __syncthreads();
    for (int s = 128; s > 0; s >>= 1) {
      if (tid < s) {
        if (rv[tid + s] > rv[tid]) { rv[tid] = rv[tid + s]; ri[tid] = ri[tid + s]; }
      }
      __syncthreads();
    }
    if (tid == 0) cur = ri[0];
    __syncthreads();
  }
}

// gather sampled coords: dst[t] = src[b, fidx[t]]
__global__ void k_gather_xyz(const float* __restrict__ src, int N,
                             const int* __restrict__ fidx, int S,
                             float* __restrict__ dst) {
  int t = blockIdx.x * 256 + threadIdx.x;
  if (t >= BSZ * S) return;
  int b = t / S;
  int j = fidx[t];
  const float* s = src + ((size_t)b * N + j) * 3;
#pragma unroll
  for (int c = 0; c < 3; ++c) dst[(size_t)t * 3 + c] = s[c];
}

// ---------------------------------------------------------------------------
// Ball query: one thread per (b,s); ascending scan == sorted-first-K semantics.
__global__ void k_query_ball(const float* __restrict__ xyz, int N,
                             const float* __restrict__ newxyz, int S, int K,
                             float r2, int* __restrict__ idx) {
  int t = blockIdx.x * 256 + threadIdx.x;
  if (t >= BSZ * S) return;
  int b = t / S;
  const float* p = xyz + (size_t)b * N * 3;
  float qx = newxyz[t * 3 + 0], qy = newxyz[t * 3 + 1], qz = newxyz[t * 3 + 2];
  int* out = idx + (size_t)t * K;
  int cnt = 0, first = 0;
  for (int n = 0; n < N && cnt < K; ++n) {
    float dx = p[n * 3 + 0] - qx;
    float dy = p[n * 3 + 1] - qy;
    float dz = p[n * 3 + 2] - qz;
    float d = dx * dx + dy * dy + dz * dz;
    if (d <= r2) { if (cnt == 0) first = n; out[cnt++] = n; }
  }
  for (; cnt < K; ++cnt) out[cnt] = first;
}

// ---------------------------------------------------------------------------
// SA1 grouping: row = (b*S+s)*K + k -> f16 [normals(3), g_xyz(3), pad..32]
__global__ void k_group_sa1(const float* __restrict__ xyz,
                            const float* __restrict__ nrm,
                            const float* __restrict__ newxyz,
                            const int* __restrict__ idx, int S, int K,
                            _Float16* __restrict__ A) {
  size_t t = (size_t)blockIdx.x * 256 + threadIdx.x;
  size_t tot = (size_t)BSZ * S * K;
  if (t >= tot) return;
  size_t bs = t / K;
  int b = (int)(bs / S);
  int j = idx[t];
  const float* pj = xyz + ((size_t)b * NPTS + j) * 3;
  const float* nj = nrm + ((size_t)b * NPTS + j) * 3;
  const float* q  = newxyz + bs * 3;
  _Float16* row = A + t * 32;
  row[0] = (_Float16)nj[0]; row[1] = (_Float16)nj[1]; row[2] = (_Float16)nj[2];
  row[3] = (_Float16)(pj[0] - q[0]);
  row[4] = (_Float16)(pj[1] - q[1]);
  row[5] = (_Float16)(pj[2] - q[2]);
#pragma unroll
  for (int c = 6; c < 32; ++c) row[c] = (_Float16)0.f;
}

// Generic grouping with features: [feats(Cf), g_xyz(3), pad..Kp]
__global__ void k_group_feats(const float* __restrict__ xyzL, int NL,
                              const float* __restrict__ F, int Cf,
                              const float* __restrict__ newxyz,
                              const int* __restrict__ idx, int S, int K,
                              _Float16* __restrict__ A, int Kp) {
  size_t t = (size_t)blockIdx.x * 256 + threadIdx.x;
  size_t tot = (size_t)BSZ * S * K;
  if (t >= tot) return;
  size_t bs = t / K;
  int b = (int)(bs / S);
  int j = idx[t];
  const float* f  = F + ((size_t)b * NL + j) * Cf;
  const float* pj = xyzL + ((size_t)b * NL + j) * 3;
  const float* q  = newxyz + bs * 3;
  _Float16* row = A + t * Kp;
  for (int c = 0; c < Cf; ++c) row[c] = (_Float16)f[c];
  row[Cf + 0] = (_Float16)(pj[0] - q[0]);
  row[Cf + 1] = (_Float16)(pj[1] - q[1]);
  row[Cf + 2] = (_Float16)(pj[2] - q[2]);
  for (int c = Cf + 3; c < Kp; ++c) row[c] = (_Float16)0.f;
}

// SA3 group-all input: rows = b*128+s; ch[0..2]=xyz2, ch[3..642]=p2, pad..672
__global__ void k_group_all(const float* __restrict__ xyz2,
                            const float* __restrict__ p2,
                            _Float16* __restrict__ A) {
  int t = blockIdx.x * 256 + threadIdx.x;
  const int ROWS = BSZ * 128, KP = 672;
  if (t >= ROWS * KP) return;
  int m = t / KP, c = t % KP;
  float v = 0.f;
  if (c < 3)        v = xyz2[(size_t)m * 3 + c];
  else if (c < 643) v = p2[(size_t)m * 640 + (c - 3)];
  A[t] = (_Float16)v;
}

// ---------------------------------------------------------------------------
// Weight conversion: W f32 [O,C] -> f16 [O,Kp] zero-padded
__global__ void k_conv_w(const float* __restrict__ W, _Float16* __restrict__ Wh,
                         int O, int C, int Kp) {
  int t = blockIdx.x * 256 + threadIdx.x;
  if (t >= O * Kp) return;
  int o = t / Kp, c = t % Kp;
  Wh[t] = (c < C) ? (_Float16)W[(size_t)o * C + c] : (_Float16)0.f;
}

__global__ void k_f32_to_f16(const float* __restrict__ X, _Float16* __restrict__ Y, int n) {
  int t = blockIdx.x * 256 + threadIdx.x;
  if (t < n) Y[t] = (_Float16)X[t];
}

// ---------------------------------------------------------------------------
// WMMA GEMM: Out[M,N] = A[M,K](f16,rowmajor) x W[N,K](f16,rowmajor)^T + bias
// K multiple of 32. 256 thr = 8 waves; tile 64x64x32; wave (wr,wc) does
// rows [wr*16,+16) x cols [wc*32,+32) via two 16x16x32 WMMAs.
// A/W tiles are streamed into LDS by the Tensor Data Mover (wave 0 issues,
// TENSORcnt tracks completion), with TDM padding producing the 40-half pitch.
__launch_bounds__(256)
__global__ void k_gemm_wmma(const _Float16* __restrict__ A,
                            const _Float16* __restrict__ W,
                            const float* __restrict__ bias,
                            float* __restrict__ Out, int M, int N, int K) {
  __shared__ __align__(16) _Float16 As[64][40];
  __shared__ __align__(16) _Float16 Ws[64][40];
  int tid  = threadIdx.x;
  int lane = tid & 31;
  int wave = tid >> 5;
  int wr = wave >> 1, wc = wave & 1;
  int m0 = blockIdx.x * 64;
  int n0 = blockIdx.y * 64;
  unsigned asOff = (unsigned)(unsigned long long)(uintptr_t)&As[0][0];
  unsigned wsOff = (unsigned)(unsigned long long)(uintptr_t)&Ws[0][0];
  v8f c0 = {}; v8f c1 = {};

  for (int k0 = 0; k0 < K; k0 += 32) {
    if (wave == 0) {
      // TDM: stream A tile (rows m0.., cols k0..k0+32) and W tile into LDS
      tdm_load_tile_f16(asOff, A + (size_t)m0 * K + k0,
                        (unsigned)(M - m0), (unsigned)(K - k0), (unsigned)K);
      tdm_load_tile_f16(wsOff, W + (size_t)n0 * K + k0,
                        (unsigned)(N - n0), (unsigned)(K - k0), (unsigned)K);
      __builtin_amdgcn_s_wait_tensorcnt(0);
    } else if (wave == 7 && k0 + 32 < K) {
      // hint the next k-step tiles toward L2 (global_prefetch_b8)
      __builtin_prefetch(A + (size_t)(m0 + lane) * K + k0 + 32, 0, 1);
      __builtin_prefetch(W + (size_t)(n0 + lane) * K + k0 + 32, 0, 1);
    }
    __syncthreads();

    // build fragments per ISA 16-bit A 16x32 layout
    v16h af, bf0, bf1;
    int mrow  = wr * 16 + (lane & 15);
    int khalf = (lane >> 4) * 8;
    int nrow  = wc * 32 + (lane & 15);
#pragma unroll
    for (int e = 0; e < 16; ++e) {
      int v  = e >> 1;
      int kk = ((v < 4) ? (2 * v) : (16 + 2 * (v - 4))) + khalf + (e & 1);
      af[e]  = As[mrow][kk];
      bf0[e] = Ws[nrow][kk];
      bf1[e] = Ws[nrow + 16][kk];
    }
    c0 = __builtin_amdgcn_wmma_f32_16x16x32_f16(false, af, false, bf0,
                                                (short)0, c0, false, false);
    c1 = __builtin_amdgcn_wmma_f32_16x16x32_f16(false, af, false, bf1,
                                                (short)0, c1, false, false);
    __syncthreads();
  }

  // store: C/D layout -> m = wr*16 + v + 8*(lane>>4), n = base + (lane&15)
  int nb = n0 + wc * 32 + (lane & 15);
  int mb = m0 + wr * 16 + ((lane >> 4) * 8);
#pragma unroll
  for (int v = 0; v < 8; ++v) {
    int gm = mb + v;
    if (gm < M) {
      if (nb < N)      Out[(size_t)gm * N + nb]      = c0[v] + (bias ? bias[nb] : 0.f);
      if (nb + 16 < N) Out[(size_t)gm * N + nb + 16] = c1[v] + (bias ? bias[nb + 16] : 0.f);
    }
  }
}

// ---------------------------------------------------------------------------
// per-channel mean/var over M rows; one block per channel; mv = [mean[N],var[N]]
__global__ void k_bn_stats(const float* __restrict__ O, int M, int N,
                           float* __restrict__ mv) {
  int c = blockIdx.x, tid = threadIdx.x;
  __shared__ float ss[256], sq[256];
  float s = 0.f, q = 0.f;
  for (int m = tid; m < M; m += 256) {
    float v = O[(size_t)m * N + c];
    s += v; q += v * v;
  }
  ss[tid] = s; sq[tid] = q;
  __syncthreads();
  for (int st = 128; st > 0; st >>= 1) {
    if (tid < st) { ss[tid] += ss[tid + st]; sq[tid] += sq[tid + st]; }
    __syncthreads();
  }
  if (tid == 0) {
    float mean = ss[0] / (float)M;
    mv[c]     = mean;
    mv[N + c] = sq[0] / (float)M - mean * mean;
  }
}

// normalize + ReLU in place; optionally emit padded f16 for next layer
__global__ void k_bn_relu(float* __restrict__ O, const float* __restrict__ mv,
                          const float* __restrict__ g, const float* __restrict__ be,
                          int M, int N, _Float16* __restrict__ Anext, int Kp) {
  size_t t = (size_t)blockIdx.x * 256 + threadIdx.x;
  if (t >= (size_t)M * Kp) return;
  int c = (int)(t % Kp);
  size_t m = t / Kp;
  if (c < N) {
    float mean = mv[c], var = mv[N + c];
    float v = O[m * N + c];
    v = g[c] * (v - mean) * rsqrtf(var + BN_EPS) + be[c];
    v = fmaxf(v, 0.f);
    O[m * N + c] = v;
    if (Anext) Anext[m * Kp + c] = (_Float16)v;
  } else if (Anext) {
    Anext[m * Kp + c] = (_Float16)0.f;
  }
}

// max over K neighbors: P[bs*Cstride+cofs+c] = max_k Out[(bs*K+k)*N+c]
__global__ void k_maxpool(const float* __restrict__ O, int rows, int K, int N,
                          float* __restrict__ P, int Cstride, int cofs) {
  int t = blockIdx.x * 256 + threadIdx.x;
  if (t >= rows * N) return;
  int bs = t / N, c = t % N;
  float m = -3.4e38f;
  for (int k = 0; k < K; ++k)
    m = fmaxf(m, O[((size_t)bs * K + k) * N + c]);
  P[(size_t)bs * Cstride + cofs + c] = m;
}

// ===========================================================================
extern "C" void kernel_launch(void* const* d_in, const int* in_sizes, int n_in,
                              void* d_out, int out_size, void* d_ws, size_t ws_size,
                              hipStream_t stream) {
  (void)in_sizes; (void)n_in; (void)out_size; (void)ws_size;
  const float* x = (const float*)d_in[0];

  static const LayerDesc SA1[3][3] = {
    {{1, 32, 6},  {5, 32, 32},  {9, 64, 32}},
    {{13, 64, 6}, {17, 64, 64}, {21, 128, 64}},
    {{25, 64, 6}, {29, 96, 64}, {33, 128, 96}},
  };
  static const LayerDesc SA2[3][3] = {
    {{37, 64, 323},  {41, 64, 64},   {45, 128, 64}},
    {{49, 128, 323}, {53, 128, 128}, {57, 256, 128}},
    {{61, 128, 323}, {65, 128, 128}, {69, 256, 128}},
  };
  static const LayerDesc SA3[3] = {{73, 256, 643}, {77, 512, 256}, {81, 1024, 512}};
  static const LayerDesc FC1 = {85, 512, 1024};
  static const LayerDesc FC2 = {89, 256, 512};
  static const LayerDesc FC3 = {93, 40, 256};   // W=93, b=94, no BN

  // --- workspace carve (bump allocator, deterministic) ---
  char* wsp = (char*)d_ws;
  auto bump = [&](size_t bytes) -> char* {
    char* r = wsp; wsp += (bytes + 255) & ~(size_t)255; return r;
  };
  float* xyz   = (float*)bump((size_t)BSZ * NPTS * 3 * 4);
  float* nrm   = (float*)bump((size_t)BSZ * NPTS * 3 * 4);
  int*   fidx1 = (int*)  bump((size_t)BSZ * 512 * 4);
  float* nx1   = (float*)bump((size_t)BSZ * 512 * 3 * 4);
  int*   fidx2 = (int*)  bump((size_t)BSZ * 128 * 4);
  float* nx2   = (float*)bump((size_t)BSZ * 128 * 3 * 4);
  int*   idxb  = (int*)  bump((size_t)BSZ * 512 * 128 * 4);
  float* p1    = (float*)bump((size_t)BSZ * 512 * 320 * 4);
  float* p2    = (float*)bump((size_t)BSZ * 128 * 640 * 4);
  float* feat  = (float*)bump((size_t)BSZ * 1024 * 4);
  float* mv    = (float*)bump((size_t)2 * 1024 * 4);
  _Float16* ping = (_Float16*)bump((size_t)524288 * 96 * 2);
  _Float16* pong = (_Float16*)bump((size_t)524288 * 96 * 2);
  float* obuf  = (float*)bump((size_t)524288 * 128 * 4);
  _Float16* wh = (_Float16*)bump((size_t)2000000 * 2);

  // --- convert all weights to padded f16 ---
  _Float16* wp = wh;
  auto convW = [&](const LayerDesc& l) -> _Float16* {
    int Kp = rnd32i(l.C);
    _Float16* dst = wp;
    wp += (size_t)l.O * Kp;
    int tot = l.O * Kp;
    k_conv_w<<<(tot + 255) / 256, 256, 0, stream>>>((const float*)d_in[l.wi],
                                                    dst, l.O, l.C, Kp);
    return dst;
  };
  _Float16* WH1[3][3]; _Float16* WH2[3][3]; _Float16* WH3[3];
  for (int i = 0; i < 3; ++i) for (int j = 0; j < 3; ++j) WH1[i][j] = convW(SA1[i][j]);
  for (int i = 0; i < 3; ++i) for (int j = 0; j < 3; ++j) WH2[i][j] = convW(SA2[i][j]);
  for (int i = 0; i < 3; ++i) WH3[i] = convW(SA3[i]);
  _Float16* WF1 = convW(FC1);
  _Float16* WF2 = convW(FC2);
  _Float16* WF3 = convW(FC3);

  // --- MLP chain runner: A0 in ping, 3x (gemm -> stats -> bn+relu) ---
  auto run_mlp3 = [&](const LayerDesc* Ls, _Float16* const* WHs, int M) -> int {
    _Float16* cur = ping; _Float16* nxt = pong;
    for (int l = 0; l < 3; ++l) {
      const LayerDesc& ly = Ls[l];
      int Kp = rnd32i(ly.C);
      dim3 g((M + 63) / 64, (ly.O + 63) / 64);
      k_gemm_wmma<<<g, 256, 0, stream>>>(cur, WHs[l],
          (const float*)d_in[ly.wi + 1], obuf, M, ly.O, Kp);
      k_bn_stats<<<ly.O, 256, 0, stream>>>(obuf, M, ly.O, mv);
      bool last = (l == 2);
      int Kpn = last ? ly.O : rnd32i(ly.O);
      size_t tot = (size_t)M * Kpn;
      k_bn_relu<<<(unsigned)((tot + 255) / 256), 256, 0, stream>>>(obuf, mv,
          (const float*)d_in[ly.wi + 2], (const float*)d_in[ly.wi + 3],
          M, ly.O, last ? (_Float16*)nullptr : nxt, Kpn);
      _Float16* t = cur; cur = nxt; nxt = t;
    }
    return Ls[2].O;
  };

  // --- split input ---
  k_split_xyz<<<(BSZ * NPTS + 255) / 256, 256, 0, stream>>>(x, xyz, nrm);

  // ================= SA1 =================
  k_fps<<<BSZ, 256, 0, stream>>>(xyz, NPTS, 512, fidx1);
  k_gather_xyz<<<(BSZ * 512 + 255) / 256, 256, 0, stream>>>(xyz, NPTS, fidx1, 512, nx1);
  {
    const float r[3] = {0.1f, 0.2f, 0.4f};
    const int  kk[3] = {16, 32, 128};
    int cofs = 0;
    for (int br = 0; br < 3; ++br) {
      int K = kk[br], S = 512;
      int M = BSZ * S * K;
      k_query_ball<<<(BSZ * S + 255) / 256, 256, 0, stream>>>(xyz, NPTS, nx1, S, K,
                                                              r[br] * r[br], idxb);
      k_group_sa1<<<(unsigned)(((size_t)M + 255) / 256), 256, 0, stream>>>(
          xyz, nrm, nx1, idxb, S, K, ping);
      int oN = run_mlp3(SA1[br], WH1[br], M);
      k_maxpool<<<(BSZ * S * oN + 255) / 256, 256, 0, stream>>>(obuf, BSZ * S, K, oN,
                                                                p1, 320, cofs);
      cofs += oN;   // 64 + 128 + 128 = 320
    }
  }

  // ================= SA2 =================
  k_fps<<<BSZ, 256, 0, stream>>>(nx1, 512, 128, fidx2);
  k_gather_xyz<<<(BSZ * 128 + 255) / 256, 256, 0, stream>>>(nx1, 512, fidx2, 128, nx2);
  {
    const float r[3] = {0.2f, 0.4f, 0.8f};
    const int  kk[3] = {32, 64, 128};
    int cofs = 0;
    for (int br = 0; br < 3; ++br) {
      int K = kk[br], S = 128;
      int M = BSZ * S * K;
      k_query_ball<<<(BSZ * S + 255) / 256, 256, 0, stream>>>(nx1, 512, nx2, S, K,
                                                              r[br] * r[br], idxb);
      k_group_feats<<<(unsigned)(((size_t)M + 255) / 256), 256, 0, stream>>>(
          nx1, 512, p1, 320, nx2, idxb, S, K, ping, 352);
      int oN = run_mlp3(SA2[br], WH2[br], M);
      k_maxpool<<<(BSZ * S * oN + 255) / 256, 256, 0, stream>>>(obuf, BSZ * S, K, oN,
                                                                p2, 640, cofs);
      cofs += oN;   // 128 + 256 + 256 = 640
    }
  }

  // ================= SA3 (group all) =================
  {
    int tot = BSZ * 128 * 672;
    k_group_all<<<(tot + 255) / 256, 256, 0, stream>>>(nx2, p2, ping);
    int oN = run_mlp3(SA3, WH3, BSZ * 128);     // -> obuf [B*128, 1024], relu'd
    // global max over the 128 points per batch: rows=B, K=128
    k_maxpool<<<(BSZ * oN + 255) / 256, 256, 0, stream>>>(obuf, BSZ, 128, oN,
                                                          feat, oN, 0);
  }

  // ================= FC head =================
  k_f32_to_f16<<<(BSZ * 1024 + 255) / 256, 256, 0, stream>>>(feat, ping, BSZ * 1024);
  // fc1: [8,1024] -> [8,512], BN + ReLU
  {
    dim3 g(1, (512 + 63) / 64);
    k_gemm_wmma<<<g, 256, 0, stream>>>(ping, WF1, (const float*)d_in[FC1.wi + 1],
                                       obuf, BSZ, 512, 1024);
    k_bn_stats<<<512, 256, 0, stream>>>(obuf, BSZ, 512, mv);
    k_bn_relu<<<(BSZ * 512 + 255) / 256, 256, 0, stream>>>(obuf, mv,
        (const float*)d_in[FC1.wi + 2], (const float*)d_in[FC1.wi + 3],
        BSZ, 512, pong, 512);
  }
  // fc2: [8,512] -> [8,256], BN + ReLU
  {
    dim3 g(1, (256 + 63) / 64);
    k_gemm_wmma<<<g, 256, 0, stream>>>(pong, WF2, (const float*)d_in[FC2.wi + 1],
                                       obuf, BSZ, 256, 512);
    k_bn_stats<<<256, 256, 0, stream>>>(obuf, BSZ, 256, mv);
    k_bn_relu<<<(BSZ * 256 + 255) / 256, 256, 0, stream>>>(obuf, mv,
        (const float*)d_in[FC2.wi + 2], (const float*)d_in[FC2.wi + 3],
        BSZ, 256, ping, 256);
  }
  // fc3: [8,256] -> [8,40] + bias, straight to d_out
  {
    dim3 g(1, 1);
    k_gemm_wmma<<<g, 256, 0, stream>>>(ping, WF3, (const float*)d_in[94],
                                       (float*)d_out, BSZ, 40, 256);
  }
}
